// Self_Attention_12378095747327
// MI455X (gfx1250) — compile-verified, matrix operand order checked by reading, not run on hardware
//
#include <hip/hip_runtime.h>

typedef __attribute__((ext_vector_type(16))) __bf16 v16bf;
typedef __attribute__((ext_vector_type(8)))  __bf16 v8bf;
typedef __attribute__((ext_vector_type(8)))  float  v8f;
typedef __attribute__((ext_vector_type(4)))  float  v4f;

#define B_    8
#define S_    2048
#define D_    512
#define NROWS (B_ * S_)      // 16384
#define QTILE 32
#define KCHUNK 64

// Combine two 8-element bf16 halves into one 16-element fragment.
static __device__ inline v16bf combine8(v8bf lo, v8bf hi) {
  return __builtin_shufflevector(lo, hi, 0,1,2,3,4,5,6,7,8,9,10,11,12,13,14,15);
}

// A/B fragment (16x32 bf16, wave32 layout): per lane the 16 elements map to
// K = (e%8) + 8*hi + 16*(e/8)  -> two contiguous 8-bf16 (16B) runs at +0 / +16.
// `base` must already point at element K = k0 + 8*hi of the lane's row/column.
static __device__ inline v16bf load_frag_bf16(const __bf16* base) {
  v8bf lo = *reinterpret_cast<const v8bf*>(base);
  v8bf hi = *reinterpret_cast<const v8bf*>(base + 16);
  return combine8(lo, hi);
}

static __device__ inline v8f wmma_bf16(v16bf a, v16bf b, v8f c) {
  return __builtin_amdgcn_wmma_f32_16x16x32_bf16(false, a, false, b, (short)0, c,
                                                 false, false);
}

// ---------------------------------------------------------------------------
// Kernel 0a: fp32 x[16384][512] -> bf16 Xb (row-major).  One-time convert so
// the GEMM hot loop has zero cvt VALU and half the load bytes.
// ---------------------------------------------------------------------------
__global__ __launch_bounds__(256)
void xconv_kernel(const float* __restrict__ x, __bf16* __restrict__ xb) {
  size_t i = ((size_t)blockIdx.x * 256 + threadIdx.x) * 8;   // 8 elems/thread
  v4f a = *reinterpret_cast<const v4f*>(x + i);
  v4f b = *reinterpret_cast<const v4f*>(x + i + 4);
  v8bf o;
#pragma unroll
  for (int r = 0; r < 4; ++r) {
    o[r]     = (__bf16)a[r];
    o[4 + r] = (__bf16)b[r];
  }
  *reinterpret_cast<v8bf*>(xb + i) = o;
}

// ---------------------------------------------------------------------------
// Kernel 0b: fp32 W[3][K=512][N=512] -> bf16 W^T[3][N=512][K=512]
// ---------------------------------------------------------------------------
__global__ __launch_bounds__(256)
void wconv_kernel(const float* __restrict__ w, __bf16* __restrict__ wt) {
  int idx = blockIdx.x * 256 + threadIdx.x;   // 3*512*512 total
  int mat = idx >> 18;
  int r   = idx & 262143;
  int k   = r >> 9;
  int n   = r & 511;
  wt[mat * 262144 + n * 512 + k] = (__bf16)w[idx];
}

// ---------------------------------------------------------------------------
// Kernel 1: QKV projection.  One wave computes a 16(M) x 64(N) tile of one of
// Q/K/V.  Q,K stored row-major bf16; V stored transposed per batch
// (Vt[b][n][s]) so the attention kernel gets contiguous B-fragments.
// ---------------------------------------------------------------------------
__global__ __launch_bounds__(256)
void qkv_kernel(const __bf16* __restrict__ Xb, const __bf16* __restrict__ Wt,
                __bf16* __restrict__ Q, __bf16* __restrict__ Kb,
                __bf16* __restrict__ Vt) {
  const int lane = threadIdx.x & 31;
  const int wave = threadIdx.x >> 5;
  const int ln = lane & 15;
  const int hi = lane >> 4;

  int tile = blockIdx.x * 8 + wave;   // 3 * 1024 * 8 = 24576 wave-tiles
  int mat  = tile >> 13;              // 0..2  (Q / K / V)
  int rem  = tile & 8191;
  int mt   = rem >> 3;                // 0..1023
  int ns   = rem & 7;                 // 0..7
  int m0 = mt * 16;
  int n0 = ns * 64;

  const __bf16* arow = Xb + (size_t)(m0 + ln) * D_;
  const __bf16* wmat = Wt + (size_t)mat * D_ * D_;

  const v8f zero = {};
  v8f acc[4];
#pragma unroll
  for (int j = 0; j < 4; ++j) acc[j] = zero;

  for (int k0 = 0; k0 < D_; k0 += 32) {
    v16bf a = load_frag_bf16(arow + k0 + 8 * hi);
#pragma unroll
    for (int j = 0; j < 4; ++j) {
      const __bf16* bcol = wmat + (size_t)(n0 + 16 * j + ln) * D_ + k0 + 8 * hi;
      acc[j] = wmma_bf16(a, load_frag_bf16(bcol), acc[j]);
    }
  }

  // D layout: lane -> column N, VGPR r -> row (M = m0 + r + 8*hi)
  if (mat < 2) {
    __bf16* dst = (mat == 0) ? Q : Kb;
#pragma unroll
    for (int j = 0; j < 4; ++j) {
      int n = n0 + 16 * j + ln;
#pragma unroll
      for (int r = 0; r < 8; ++r) {
        int m = m0 + r + 8 * hi;
        dst[(size_t)m * D_ + n] = (__bf16)acc[j][r];
      }
    }
  } else {
    int b  = m0 >> 11;        // batch
    int s0 = m0 & 2047;       // sequence pos within batch
#pragma unroll
    for (int j = 0; j < 4; ++j) {
      int n = n0 + 16 * j + ln;
      v8bf pack;
#pragma unroll
      for (int r = 0; r < 8; ++r) pack[r] = (__bf16)acc[j][r];
      *reinterpret_cast<v8bf*>(Vt + (size_t)b * D_ * S_ + (size_t)n * S_ + s0 +
                               8 * hi) = pack;
    }
  }
}

// ---------------------------------------------------------------------------
// Kernel 2: flash attention.  Block = 8 waves, 32 query rows, 64-key chunks.
// Wave w: mi = w/4 (16-row half of Q tile), nj = w%4.
//   scores: wave computes S[mi][nj] (16x16) via 16 bf16 WMMAs (K=512)
//   softmax: 8 threads per row (shfl_xor reductions), P -> LDS bf16
//   PV: wave owns O tiles (mi, n-tiles nj*8..nj*8+7), 16 WMMAs per chunk
// ---------------------------------------------------------------------------
__global__ __launch_bounds__(256)
void flash_kernel(const __bf16* __restrict__ Q, const __bf16* __restrict__ Kb,
                  const __bf16* __restrict__ Vt, float* __restrict__ out) {
  __shared__ alignas(16) float  s_tile[QTILE * KCHUNK];
  __shared__ alignas(16) __bf16 p_tile[QTILE * KCHUNK];
  __shared__ float row_m[QTILE];
  __shared__ float row_l[QTILE];
  __shared__ float row_a[QTILE];

  const int tid  = threadIdx.x;
  const int lane = tid & 31;
  const int wave = tid >> 5;
  const int ln = lane & 15;
  const int hi = lane >> 4;
  const int mi = wave >> 2;   // 0..1
  const int nj = wave & 3;    // 0..3

  const int b  = blockIdx.y;
  const int q0 = blockIdx.x * QTILE;

  const float scale = 0.044194173824159216f;   // 1/sqrt(512)

  // Preload this wave's Q fragments (16 rows x full depth, 16 k-chunks).
  const __bf16* qrow = Q + (size_t)(b * S_ + q0 + mi * 16 + ln) * D_;
  v16bf qf[16];
#pragma unroll
  for (int c = 0; c < 16; ++c) qf[c] = load_frag_bf16(qrow + c * 32 + 8 * hi);

  const v8f zero = {};
  v8f oacc[8];
#pragma unroll
  for (int t = 0; t < 8; ++t) oacc[t] = zero;

  if (tid < QTILE) { row_m[tid] = -3.0e38f; row_l[tid] = 0.0f; }
  __syncthreads();

  const __bf16* kbase = Kb + (size_t)b * S_ * D_;
  const __bf16* vbase = Vt + (size_t)b * D_ * S_;

  for (int j0 = 0; j0 < S_; j0 += KCHUNK) {
    // Pull next chunk's K/V lines toward the WGP while we compute this one.
    if (j0 + KCHUNK < S_) {
      __builtin_prefetch(kbase + (size_t)(j0 + KCHUNK + nj * 16 + ln) * D_, 0, 0);
      __builtin_prefetch(vbase + (size_t)((nj * 8) * 16 + ln) * S_ + j0 + KCHUNK,
                         0, 0);
    }

    // ---- scores: S = Q K^T  (B-fragment = K rows, contiguous in depth)
    v8f sacc = zero;
#pragma unroll
    for (int c = 0; c < 16; ++c) {
      const __bf16* kcol =
          kbase + (size_t)(j0 + nj * 16 + ln) * D_ + c * 32 + 8 * hi;
      sacc = wmma_bf16(qf[c], load_frag_bf16(kcol), sacc);
    }
#pragma unroll
    for (int r = 0; r < 8; ++r) {
      int ml = mi * 16 + r + 8 * hi;
      s_tile[ml * KCHUNK + nj * 16 + ln] = sacc[r] * scale;
    }
    __syncthreads();

    // ---- online softmax: 8 threads per row, shfl_xor reductions (same wave)
    {
      const int r  = tid >> 3;       // row 0..31
      const int sc = (tid & 7) * 8;  // this thread's 8-column slice
      float vals[8];
      float mmax = -3.0e38f;
#pragma unroll
      for (int c = 0; c < 8; ++c) {
        vals[c] = s_tile[r * KCHUNK + sc + c];
        mmax = fmaxf(mmax, vals[c]);
      }
#pragma unroll
      for (int m = 1; m < 8; m <<= 1)
        mmax = fmaxf(mmax, __shfl_xor(mmax, m, 8));
      float mold = row_m[r];                 // read before lane-0 write (same wave)
      float mnew = fmaxf(mold, mmax);
      float lpart = 0.0f;
#pragma unroll
      for (int c = 0; c < 8; ++c) {
        float p = __expf(vals[c] - mnew);
        lpart += p;
        p_tile[r * KCHUNK + sc + c] = (__bf16)p;
      }
#pragma unroll
      for (int m = 1; m < 8; m <<= 1)
        lpart += __shfl_xor(lpart, m, 8);
      if ((tid & 7) == 0) {
        float alpha = __expf(mold - mnew);
        row_a[r] = alpha;
        row_l[r] = row_l[r] * alpha + lpart;
        row_m[r] = mnew;
      }
    }
    __syncthreads();

    // ---- rescale O, then O += P * V_chunk
    float av[8];
#pragma unroll
    for (int r = 0; r < 8; ++r) av[r] = row_a[mi * 16 + r + 8 * hi];
#pragma unroll
    for (int t = 0; t < 8; ++t)
#pragma unroll
      for (int r = 0; r < 8; ++r) oacc[t][r] *= av[r];

#pragma unroll
    for (int kk = 0; kk < 2; ++kk) {
      v16bf pf =
          load_frag_bf16(&p_tile[(mi * 16 + ln) * KCHUNK + kk * 32 + 8 * hi]);
#pragma unroll
      for (int t = 0; t < 8; ++t) {
        int n = (nj * 8 + t) * 16 + ln;
        const __bf16* vcol = vbase + (size_t)n * S_ + j0 + kk * 32 + 8 * hi;
        oacc[t] = wmma_bf16(pf, load_frag_bf16(vcol), oacc[t]);
      }
    }
    // next iteration's s_tile write is fenced by the sync after softmax;
    // p_tile rewrite is fenced by next iteration's first sync.
  }

  // ---- final normalize + store fp32
  float rl[8];
#pragma unroll
  for (int r = 0; r < 8; ++r) rl[r] = 1.0f / row_l[mi * 16 + r + 8 * hi];
#pragma unroll
  for (int t = 0; t < 8; ++t) {
    int n = (nj * 8 + t) * 16 + ln;
#pragma unroll
    for (int r = 0; r < 8; ++r) {
      int m = q0 + mi * 16 + r + 8 * hi;
      out[((size_t)b * S_ + m) * D_ + n] = oacc[t][r] * rl[r];
    }
  }
}

// ---------------------------------------------------------------------------
extern "C" void kernel_launch(void* const* d_in, const int* in_sizes, int n_in,
                              void* d_out, int out_size, void* d_ws,
                              size_t ws_size, hipStream_t stream) {
  const float* x = (const float*)d_in[0];      // (8, 2048, 512) fp32
  const float* w = (const float*)d_in[1];      // (3, 512, 512) fp32
  float* out = (float*)d_out;                  // (8, 2048, 512) fp32

  char* ws = (char*)d_ws;
  __bf16* Wt = (__bf16*)ws;                                   // 1.5 MB
  __bf16* Xb = (__bf16*)(ws + (size_t)3 * D_ * D_ * 2);       // 16 MB
  __bf16* Qb = Xb + (size_t)NROWS * D_;                       // 16 MB
  __bf16* Kb = Qb + (size_t)NROWS * D_;                       // 16 MB
  __bf16* Vt = Kb + (size_t)NROWS * D_;                       // 16 MB

  xconv_kernel<<<NROWS * D_ / (256 * 8), 256, 0, stream>>>(x, Xb);  // 4096 blocks
  wconv_kernel<<<3072, 256, 0, stream>>>(w, Wt);
  qkv_kernel<<<3072, 256, 0, stream>>>(Xb, Wt, Qb, Kb, Vt);
  dim3 grid(S_ / QTILE, B_);   // (64, 8)
  flash_kernel<<<grid, 256, 0, stream>>>(Qb, Kb, Vt, out);
}
